// Model_88459146428520
// MI455X (gfx1250) — matrix operand order repaired and checked by reading, not compiled, not run on hardware
//
#include <hip/hip_runtime.h>
#include <math.h>

// ---------------------------------------------------------------------------
// CDNA5 WMMA helpers (gfx1250, wave32). D = A(16x32 f16) * B(32x16 f16) + C(f32)
// A/B fragments use contiguous K = 16*(lane>>4) + elem, same order in both operands.
// ---------------------------------------------------------------------------
typedef __attribute__((ext_vector_type(16))) _Float16 v16h;
typedef __attribute__((ext_vector_type(8)))  _Float16 v8h;
typedef __attribute__((ext_vector_type(8)))  float    v8f;

// A-matrix fragment 16x32 f16 from LDS (row-major activations, pitch in elems).
// lane L (0-15): M=L, K = g*16+{0..15} with g = L>>4.
// Both chunks are contiguous -> two 16B LDS loads.
__device__ __forceinline__ v16h lds_afrag(const _Float16* lds, int pitch,
                                          int row0, int k0, int lane) {
    const int m = row0 + (lane & 15);
    const int g = (lane >> 4) & 1;
    const _Float16* rp = lds + m * pitch + k0 + g * 16;
    v8h lo = *(const v8h*)rp;          // K = g*16 + 0..7       (elements 0..7)
    v8h hi = *(const v8h*)(rp + 8);    // K = g*16 + 8..15      (elements 8..15)
    v16h a;
#pragma unroll
    for (int j = 0; j < 8; ++j) { a[j] = lo[j]; a[8 + j] = hi[j]; }
    return a;
}

// B-matrix fragment 32x16 f16 from LDS staged TRANSPOSED (N-major, [N][Kpad]).
// lane L: N = n0+(L&15), K = k0 + (L>>4)*16 + {0..15}  -> one contiguous 32B load.
__device__ __forceinline__ v16h lds_bfragT(const _Float16* ldsT, int kpitch,
                                           int k0, int n0, int lane) {
    const int n = n0 + (lane & 15);
    const int g = (lane >> 4) & 1;
    return *(const v16h*)(ldsT + n * kpitch + k0 + g * 16);
}

#define WMMA_F16(acc, a, b) \
    __builtin_amdgcn_wmma_f32_16x16x32_f16(false, (a), false, (b), (short)0, (acc), false, false)

// C/D 16x16 f32: VGPR r -> lanes 0-15 M=r, lanes 16-31 M=r+8; N = lane&15
#define C_ROW_BASE(rt, lane)  ((rt) * 16 + (((lane) >> 4) << 3))
#define C_COL(ct, lane)       ((ct) * 16 + ((lane) & 15))

// ---------------------------------------------------------------------------
// Problem constants
// ---------------------------------------------------------------------------
#define NPTS   16384
#define NIN    32
#define NH     64
#define KNB    32          // neighbors per query
#define D1     68          // NH + 4
#define D1P    96          // padded to 3 K-steps of 32
#define D2     128         // FP hidden

// ===========================================================================
// 1) lin_in: h = relu(relu(x@W1+b1)@W2+b2)      x:[N,32] -> h:[N,64]
//    Input tile staged with gfx1250 async global->LDS DMA (ASYNCcnt path).
// ===========================================================================
__global__ __launch_bounds__(128) void lin_in_kernel(
    const float* __restrict__ x, const float* __restrict__ W1,
    const float* __restrict__ b1, const float* __restrict__ W2,
    const float* __restrict__ b2, float* __restrict__ h) {
    __shared__ __align__(16) float    sx32[16 * 32];   // raw f32 tile (async DMA dest)
    __shared__ __align__(32) _Float16 sx[16 * 32];
    __shared__ __align__(32) _Float16 sw1T[64 * 32];   // W1^T : [N=64][K=32]
    __shared__ __align__(32) _Float16 st1[16 * 64];
    __shared__ __align__(32) _Float16 sw2T[64 * 64];   // W2^T : [N=64][K=64]
    __shared__ float sb1[64], sb2[64];

    const int tid  = threadIdx.x;
    const int lane = tid & 31;
    const int wave = tid >> 5;          // 4 waves, one 16-col tile each
    const int row0 = blockIdx.x * 16;

    // ---- async DMA: 16 rows x 32 f32 = 2 KB, one b128 per thread
    {
        const float* gsrc = x + (size_t)row0 * 32 + tid * 4;
        unsigned int loff = (unsigned int)(uintptr_t)(&sx32[tid * 4]);
        asm volatile("global_load_async_to_lds_b128 %0, %1, off"
                     :: "v"(loff), "v"(gsrc) : "memory");
    }
    // ---- stage weights transposed while DMA is in flight
    for (int t = tid; t < 64 * 32; t += 128) {          // n = t>>5, k = t&31
        sw1T[t] = (_Float16)W1[(t & 31) * 64 + (t >> 5)];
    }
    for (int t = tid; t < 64 * 64; t += 128) {          // n = t>>6, k = t&63
        sw2T[t] = (_Float16)W2[(t & 63) * 64 + (t >> 6)];
    }
    if (tid < 64) { sb1[tid] = b1[tid]; sb2[tid] = b2[tid]; }
    asm volatile("s_wait_asynccnt 0x0" ::: "memory");
    __syncthreads();
    // convert the DMA'd tile to f16
    for (int t = tid; t < 16 * 32; t += 128) sx[t] = (_Float16)sx32[t];
    __syncthreads();

    // layer 1: K = 32 (one WMMA per tile)
    {
        v8f acc = {};
        v16h a = lds_afrag(sx, 32, 0, 0, lane);
        v16h b = lds_bfragT(sw1T, 32, 0, wave * 16, lane);
        acc = WMMA_F16(acc, a, b);
        const int n  = C_COL(wave, lane);
        const int mb = C_ROW_BASE(0, lane);
        const float bv = sb1[n];
#pragma unroll
        for (int r = 0; r < 8; ++r)
            st1[(mb + r) * 64 + n] = (_Float16)fmaxf(acc[r] + bv, 0.f);
    }
    __syncthreads();

    // layer 2: K = 64 (two WMMAs)
    {
        v8f acc = {};
#pragma unroll
        for (int ks = 0; ks < 2; ++ks) {
            v16h a = lds_afrag(st1, 64, 0, ks * 32, lane);
            v16h b = lds_bfragT(sw2T, 64, ks * 32, wave * 16, lane);
            acc = WMMA_F16(acc, a, b);
        }
        const int n  = C_COL(wave, lane);
        const int mb = C_ROW_BASE(0, lane);
        const float bv = sb2[n];
#pragma unroll
        for (int r = 0; r < 8; ++r)
            h[(size_t)(row0 + mb + r) * 64 + n] = fmaxf(acc[r] + bv, 0.f);
    }
}

// ===========================================================================
// 2) FPS: sequential farthest point sampling, single block, start idx 0.
// ===========================================================================
__global__ __launch_bounds__(256) void fps_kernel(
    const float* __restrict__ pos, int Np, int ns,
    int* __restrict__ idx, float* __restrict__ dmin) {
    __shared__ float rv[256];
    __shared__ int   ri[256];
    __shared__ int   s_last;

    const int tid = threadIdx.x;
    for (int i = tid; i < Np; i += 256) dmin[i] = 3.4e38f;
    if (tid == 0) { idx[0] = 0; s_last = 0; }
    __syncthreads();

    for (int s = 1; s < ns; ++s) {
        const int last = s_last;
        const float lx = pos[last * 3 + 0];
        const float ly = pos[last * 3 + 1];
        const float lz = pos[last * 3 + 2];
        float best = -1.f; int bi = 0x7fffffff;
        for (int i = tid; i < Np; i += 256) {
            const float dx = pos[i * 3 + 0] - lx;
            const float dy = pos[i * 3 + 1] - ly;
            const float dz = pos[i * 3 + 2] - lz;
            float dm = fminf(dmin[i], dx * dx + dy * dy + dz * dz);
            dmin[i] = dm;
            if (dm > best || (dm == best && i < bi)) { best = dm; bi = i; }
        }
        rv[tid] = best; ri[tid] = bi;
        __syncthreads();
        for (int o = 128; o > 0; o >>= 1) {
            if (tid < o) {
                if (rv[tid + o] > rv[tid] ||
                    (rv[tid + o] == rv[tid] && ri[tid + o] < ri[tid])) {
                    rv[tid] = rv[tid + o]; ri[tid] = ri[tid + o];
                }
            }
            __syncthreads();
        }
        if (tid == 0) { idx[s] = ri[0]; s_last = ri[0]; }
        __syncthreads();
    }
}

// gather pos/norm for sampled indices
__global__ void gather_pn_kernel(const float* __restrict__ p, const float* __restrict__ n,
                                 const int* __restrict__ idx, int M,
                                 float* __restrict__ pq, float* __restrict__ nq) {
    int m = blockIdx.x * blockDim.x + threadIdx.x;
    if (m >= M) return;
    int j = idx[m];
#pragma unroll
    for (int c = 0; c < 3; ++c) {
        pq[m * 3 + c] = p[j * 3 + c];
        nq[m * 3 + c] = n[j * 3 + c];
    }
}

// ===========================================================================
// 3) radius-bounded 32-NN (stable ascending d2; -1 = invalid)
// ===========================================================================
__global__ void knn_radius_kernel(const float* __restrict__ pq, int M,
                                  const float* __restrict__ p, int Np,
                                  float r2, int* __restrict__ nb) {
    int m = blockIdx.x * blockDim.x + threadIdx.x;
    if (m >= M) return;
    const float qx = pq[m * 3 + 0], qy = pq[m * 3 + 1], qz = pq[m * 3 + 2];
    float db[KNB]; int ib[KNB]; int cnt = 0;
    for (int j = 0; j < Np; ++j) {
        const float dx = p[j * 3 + 0] - qx;
        const float dy = p[j * 3 + 1] - qy;
        const float dz = p[j * 3 + 2] - qz;
        const float d2 = dx * dx + dy * dy + dz * dz;
        if (d2 > r2) continue;
        if (cnt == KNB && d2 >= db[KNB - 1]) continue;
        int pos = (cnt < KNB) ? cnt : (KNB - 1);
        while (pos > 0 && db[pos - 1] > d2) {
            db[pos] = db[pos - 1]; ib[pos] = ib[pos - 1]; --pos;
        }
        db[pos] = d2; ib[pos] = j;
        if (cnt < KNB) ++cnt;
    }
    for (int k = 0; k < KNB; ++k) nb[(size_t)m * KNB + k] = (k < cnt) ? ib[k] : -1;
}

// ===========================================================================
// 4) fused SA kernel: edge features (x_j || PPF) -> relu MLP(68,68) x2 ->
//    masked max over 32 neighbors -> relu( . @ W3[68x64] + b3 )
//    2 queries per block (64 rows), 128 threads = 4 waves.
// ===========================================================================
__global__ __launch_bounds__(128) void sa_kernel(
    const float* __restrict__ hsrc, const float* __restrict__ psrc,
    const float* __restrict__ nsrc,
    const float* __restrict__ pq, const float* __restrict__ nq, int M,
    const int* __restrict__ nb,
    const float* __restrict__ W1, const float* __restrict__ B1,
    const float* __restrict__ W2, const float* __restrict__ B2,
    const float* __restrict__ W3, const float* __restrict__ B3,
    float* __restrict__ hq_out) {
    __shared__ __align__(32) _Float16 fA[64 * D1P];    // 12 KB edge feats / L2 out
    __shared__ __align__(32) _Float16 fB[64 * D1P];    // 12 KB layer-1 out
    __shared__ __align__(32) _Float16 wtT[D1P * D1P];  // 18 KB weight^T [N][K]
    __shared__ float    sbias[D1P];
    __shared__ float    sagg[2 * D1];
    __shared__ unsigned char maskv[64];

    const int tid  = threadIdx.x;
    const int lane = tid & 31;
    const int wave = tid >> 5;
    const int q0   = blockIdx.x * 2;

    __builtin_prefetch(W2 + tid, 0, 1);
    __builtin_prefetch(W3 + tid, 0, 1);

    // ---- build edge features: one (query, neighbor) row per thread (tid<64)
    if (tid < 64) {
        const int ql = tid >> 5;             // 0/1
        const int k  = tid & 31;
        const int q  = q0 + ql;
        const int r  = tid;
        int nbi = (q < M) ? nb[(size_t)q * KNB + k] : -1;
        const int valid = nbi >= 0;
        const int j = valid ? nbi : 0;
        maskv[r] = (unsigned char)valid;

        const float* xj = hsrc + (size_t)j * NH;
        for (int c = 0; c < NH; ++c) fA[r * D1P + c] = (_Float16)xj[c];

        const float qx = pq[q * 3 + 0], qy = pq[q * 3 + 1], qz = pq[q * 3 + 2];
        const float ax = nq[q * 3 + 0], ay = nq[q * 3 + 1], az = nq[q * 3 + 2];
        const float bx = nsrc[j * 3 + 0], by = nsrc[j * 3 + 1], bz = nsrc[j * 3 + 2];
        const float px = psrc[j * 3 + 0] - qx;
        const float py = psrc[j * 3 + 1] - qy;
        const float pz = psrc[j * 3 + 2] - qz;

        const float pn = sqrtf(fmaxf(px * px + py * py + pz * pz, 1e-12f));
        // angle(a,b) = atan2(||a x b||, a.b)
        float cx = ay * pz - az * py, cy = az * px - ax * pz, cz = ax * py - ay * px;
        const float a1 = atan2f(sqrtf(fmaxf(cx * cx + cy * cy + cz * cz, 1e-12f)),
                                ax * px + ay * py + az * pz);
        cx = by * pz - bz * py; cy = bz * px - bx * pz; cz = bx * py - by * px;
        const float a2 = atan2f(sqrtf(fmaxf(cx * cx + cy * cy + cz * cz, 1e-12f)),
                                bx * px + by * py + bz * pz);
        cx = ay * bz - az * by; cy = az * bx - ax * bz; cz = ax * by - ay * bx;
        const float a3 = atan2f(sqrtf(fmaxf(cx * cx + cy * cy + cz * cz, 1e-12f)),
                                ax * bx + ay * by + az * bz);
        fA[r * D1P + 64] = (_Float16)pn;
        fA[r * D1P + 65] = (_Float16)a1;
        fA[r * D1P + 66] = (_Float16)a2;
        fA[r * D1P + 67] = (_Float16)a3;
        for (int c = D1; c < D1P; ++c) fA[r * D1P + c] = (_Float16)0.f;
    }
    // ---- stage W1^T (zero-padded to 96x96) + bias    wtT[n*96+k] = W1[k*68+n]
    for (int t = tid; t < D1P * D1P; t += 128) {
        int n = t / D1P, k = t % D1P;
        wtT[t] = (n < D1 && k < D1) ? (_Float16)W1[k * D1 + n] : (_Float16)0.f;
    }
    if (tid < D1P) sbias[tid] = (tid < D1) ? B1[tid] : 0.f;
    __syncthreads();

    // ---- layer 1: 4 row tiles x 6 col tiles, 3 K-steps each
    for (int t = wave; t < 24; t += 4) {
        const int rt = t / 6, ct = t % 6;
        v8f acc = {};
#pragma unroll
        for (int ks = 0; ks < 3; ++ks) {
            v16h a = lds_afrag(fA, D1P, rt * 16, ks * 32, lane);
            v16h b = lds_bfragT(wtT, D1P, ks * 32, ct * 16, lane);
            acc = WMMA_F16(acc, a, b);
        }
        const int n  = C_COL(ct, lane);
        const int mb = C_ROW_BASE(rt, lane);
        const float bv = sbias[n];
#pragma unroll
        for (int r = 0; r < 8; ++r)
            fB[(mb + r) * D1P + n] = (_Float16)fmaxf(acc[r] + bv, 0.f);
    }
    __syncthreads();

    // ---- stage W2^T + bias
    for (int t = tid; t < D1P * D1P; t += 128) {
        int n = t / D1P, k = t % D1P;
        wtT[t] = (n < D1 && k < D1) ? (_Float16)W2[k * D1 + n] : (_Float16)0.f;
    }
    if (tid < D1P) sbias[tid] = (tid < D1) ? B2[tid] : 0.f;
    __syncthreads();

    // ---- layer 2 (output overwrites fA)
    for (int t = wave; t < 24; t += 4) {
        const int rt = t / 6, ct = t % 6;
        v8f acc = {};
#pragma unroll
        for (int ks = 0; ks < 3; ++ks) {
            v16h a = lds_afrag(fB, D1P, rt * 16, ks * 32, lane);
            v16h b = lds_bfragT(wtT, D1P, ks * 32, ct * 16, lane);
            acc = WMMA_F16(acc, a, b);
        }
        const int n  = C_COL(ct, lane);
        const int mb = C_ROW_BASE(rt, lane);
        const float bv = sbias[n];
#pragma unroll
        for (int r = 0; r < 8; ++r)
            fA[(mb + r) * D1P + n] = (_Float16)fmaxf(acc[r] + bv, 0.f);
    }
    __syncthreads();

    // ---- masked max over 32 neighbors (2 queries x 68 features)
    if (tid < 2 * D1) {
        const int ql = tid / D1, c = tid % D1;
        float vmax = -3.4e38f;
        for (int k = 0; k < KNB; ++k) {
            const int r = ql * 32 + k;
            if (maskv[r]) vmax = fmaxf(vmax, (float)fA[r * D1P + c]);
        }
        sagg[ql * D1 + c] = vmax;
    }
    __syncthreads();

    // ---- hq = relu(agg @ W3 + b3)   (2 x 64 outputs, one per thread)
    {
        const int ql = tid >> 6, n = tid & 63;
        float s = 0.f;
        for (int c = 0; c < D1; ++c) s += sagg[ql * D1 + c] * W3[c * NH + n];
        const int q = q0 + ql;
        if (q < M) hq_out[(size_t)q * NH + n] = fmaxf(s + B3[n], 0.f);
    }
}

// ===========================================================================
// 5) FP: 3-NN inverse-sq-dist interpolation + concat -> cat[Nf,128]
// ===========================================================================
__global__ void fp_interp_kernel(const float* __restrict__ ps, int Nf,
                                 const float* __restrict__ pc, int Nc,
                                 const float* __restrict__ hc,   // [Nc,64]
                                 const float* __restrict__ hs,   // [Nf,64]
                                 float* __restrict__ cat) {      // [Nf,128]
    int f = blockIdx.x * blockDim.x + threadIdx.x;
    if (f >= Nf) return;
    const float qx = ps[f * 3 + 0], qy = ps[f * 3 + 1], qz = ps[f * 3 + 2];
    float d0 = 3.4e38f, d1 = 3.4e38f, d2 = 3.4e38f;
    int i0 = 0, i1 = 0, i2 = 0;
    for (int j = 0; j < Nc; ++j) {
        const float dx = pc[j * 3 + 0] - qx;
        const float dy = pc[j * 3 + 1] - qy;
        const float dz = pc[j * 3 + 2] - qz;
        const float d = dx * dx + dy * dy + dz * dz;
        if (d < d0)      { d2 = d1; i2 = i1; d1 = d0; i1 = i0; d0 = d; i0 = j; }
        else if (d < d1) { d2 = d1; i2 = i1; d1 = d; i1 = j; }
        else if (d < d2) { d2 = d; i2 = j; }
    }
    const float w0 = 1.f / fmaxf(d0, 1e-16f);
    const float w1 = 1.f / fmaxf(d1, 1e-16f);
    const float w2 = 1.f / fmaxf(d2, 1e-16f);
    const float inv = 1.f / (w0 + w1 + w2);
    const float* h0 = hc + (size_t)i0 * NH;
    const float* h1 = hc + (size_t)i1 * NH;
    const float* h2 = hc + (size_t)i2 * NH;
    float* dst = cat + (size_t)f * D2;
    for (int c = 0; c < NH; ++c)
        dst[c] = (w0 * h0[c] + w1 * h1[c] + w2 * h2[c]) * inv;
    const float* hsr = hs + (size_t)f * NH;
    for (int c = 0; c < NH; ++c) dst[NH + c] = hsr[c];
}

// ===========================================================================
// 6) FP MLP: relu(relu(cat@W1+b1)@W2+b2)   [Nf,128] -> [Nf,64]
// ===========================================================================
__global__ __launch_bounds__(128) void fp_mlp_kernel(
    const float* __restrict__ cat, const float* __restrict__ W1,
    const float* __restrict__ b1, const float* __restrict__ W2,
    const float* __restrict__ b2, float* __restrict__ out, int Nf) {
    __shared__ __align__(32) _Float16 sin_[16 * 128];   // 4 KB
    __shared__ __align__(32) _Float16 sw1T[128 * 128];  // 32 KB  W1^T [N=128][K=128]
    __shared__ __align__(32) _Float16 st1[16 * 128];    // 4 KB
    __shared__ __align__(32) _Float16 sw2T[64 * 128];   // 16 KB  W2^T [N=64][K=128]
    __shared__ float sb1[128], sb2[64];

    const int tid  = threadIdx.x;
    const int lane = tid & 31;
    const int wave = tid >> 5;
    const int row0 = blockIdx.x * 16;

    __builtin_prefetch(W2 + tid, 0, 1);
    for (int t = tid; t < 16 * 128; t += 128)  sin_[t] = (_Float16)cat[(size_t)row0 * 128 + t];
    for (int t = tid; t < 128 * 128; t += 128) {         // n = t>>7, k = t&127
        sw1T[t] = (_Float16)W1[(t & 127) * 128 + (t >> 7)];
    }
    for (int t = tid; t < 64 * 128; t += 128) {          // n = t>>7, k = t&127
        sw2T[t] = (_Float16)W2[(t & 127) * 64 + (t >> 7)];
    }
    if (tid < 128) sb1[tid] = b1[tid];
    if (tid < 64)  sb2[tid] = b2[tid];
    __syncthreads();

    // layer 1: 8 col tiles (wave, wave+4), K = 128 -> 4 steps
    for (int ct = wave; ct < 8; ct += 4) {
        v8f acc = {};
#pragma unroll
        for (int ks = 0; ks < 4; ++ks) {
            v16h a = lds_afrag(sin_, 128, 0, ks * 32, lane);
            v16h b = lds_bfragT(sw1T, 128, ks * 32, ct * 16, lane);
            acc = WMMA_F16(acc, a, b);
        }
        const int n  = C_COL(ct, lane);
        const int mb = C_ROW_BASE(0, lane);
        const float bv = sb1[n];
#pragma unroll
        for (int r = 0; r < 8; ++r)
            st1[(mb + r) * 128 + n] = (_Float16)fmaxf(acc[r] + bv, 0.f);
    }
    __syncthreads();

    // layer 2: 4 col tiles (ct == wave), K = 128 -> 4 steps
    {
        const int ct = wave;
        v8f acc = {};
#pragma unroll
        for (int ks = 0; ks < 4; ++ks) {
            v16h a = lds_afrag(st1, 128, 0, ks * 32, lane);
            v16h b = lds_bfragT(sw2T, 128, ks * 32, ct * 16, lane);
            acc = WMMA_F16(acc, a, b);
        }
        const int n  = C_COL(ct, lane);
        const int mb = C_ROW_BASE(0, lane);
        const float bv = sb2[n];
#pragma unroll
        for (int r = 0; r < 8; ++r)
            out[(size_t)(row0 + mb + r) * 64 + n] = fmaxf(acc[r] + bv, 0.f);
    }
}

// ===========================================================================
// 7) head 1: x1 = relu(hf@W1+b1)@W2+b2        [N,64] -> [N,2]
// ===========================================================================
__global__ __launch_bounds__(128) void head1_kernel(
    const float* __restrict__ hf, const float* __restrict__ W1,
    const float* __restrict__ b1, const float* __restrict__ W2,
    const float* __restrict__ b2, float* __restrict__ out) {
    __shared__ __align__(32) _Float16 sin_[16 * 64];
    __shared__ __align__(32) _Float16 sw1T[64 * 64];    // W1^T [N=64][K=64]
    __shared__ float st1[16 * 64];
    __shared__ float sb1[64];

    const int tid  = threadIdx.x;
    const int lane = tid & 31;
    const int wave = tid >> 5;
    const int row0 = blockIdx.x * 16;

    for (int t = tid; t < 16 * 64; t += 128) sin_[t] = (_Float16)hf[(size_t)row0 * 64 + t];
    for (int t = tid; t < 64 * 64; t += 128) {           // n = t>>6, k = t&63
        sw1T[t] = (_Float16)W1[(t & 63) * 64 + (t >> 6)];
    }
    if (tid < 64) sb1[tid] = b1[tid];
    __syncthreads();

    {
        v8f acc = {};
#pragma unroll
        for (int ks = 0; ks < 2; ++ks) {
            v16h a = lds_afrag(sin_, 64, 0, ks * 32, lane);
            v16h b = lds_bfragT(sw1T, 64, ks * 32, wave * 16, lane);
            acc = WMMA_F16(acc, a, b);
        }
        const int n  = C_COL(wave, lane);
        const int mb = C_ROW_BASE(0, lane);
        const float bv = sb1[n];
#pragma unroll
        for (int r = 0; r < 8; ++r)
            st1[(mb + r) * 64 + n] = fmaxf(acc[r] + bv, 0.f);
    }
    __syncthreads();

    if (tid < 32) {
        const int r = tid >> 1, c = tid & 1;
        float s = 0.f;
        for (int n = 0; n < 64; ++n) s += st1[r * 64 + n] * W2[n * 2 + c];
        out[(size_t)(row0 + r) * 2 + c] = s + b2[c];
    }
}

// ===========================================================================
// 8) head 2: g = mean(h3); x2 = relu(g@W1+b1)@W2+b2     -> 3 floats
// ===========================================================================
__global__ __launch_bounds__(64) void head2_kernel(
    const float* __restrict__ h3, const float* __restrict__ W1,
    const float* __restrict__ b1, const float* __restrict__ W2,
    const float* __restrict__ b2, float* __restrict__ out) {
    __shared__ float g[64];
    __shared__ float t[64];
    const int n = threadIdx.x;
    float s = 0.f;
    for (int r = 0; r < 256; ++r) s += h3[(size_t)r * 64 + n];
    g[n] = s * (1.f / 256.f);
    __syncthreads();
    float a = 0.f;
    for (int c = 0; c < 64; ++c) a += g[c] * W1[c * 64 + n];
    t[n] = fmaxf(a + b1[n], 0.f);
    __syncthreads();
    if (n < 3) {
        float o = 0.f;
        for (int c = 0; c < 64; ++c) o += t[c] * W2[c * 3 + n];
        out[n] = o + b2[n];
    }
}

// ===========================================================================
// driver
// ===========================================================================
extern "C" void kernel_launch(void* const* d_in, const int* in_sizes, int n_in,
                              void* d_out, int out_size, void* d_ws, size_t ws_size,
                              hipStream_t stream) {
    (void)in_sizes; (void)n_in; (void)out_size; (void)ws_size;
    const float* x      = (const float*)d_in[0];
    const float* pos    = (const float*)d_in[1];
    const float* norm   = (const float*)d_in[2];
    const float* li_W1  = (const float*)d_in[3];
    const float* li_b1  = (const float*)d_in[4];
    const float* li_W2  = (const float*)d_in[5];
    const float* li_b2  = (const float*)d_in[6];
    const float* sa1W1  = (const float*)d_in[7];
    const float* sa1b1  = (const float*)d_in[8];
    const float* sa1W2  = (const float*)d_in[9];
    const float* sa1b2  = (const float*)d_in[10];
    const float* sa2W   = (const float*)d_in[11];
    const float* sa2b   = (const float*)d_in[12];
    const float* fpW1   = (const float*)d_in[13];
    const float* fpb1   = (const float*)d_in[14];
    const float* fpW2   = (const float*)d_in[15];
    const float* fpb2   = (const float*)d_in[16];
    const float* o1W1   = (const float*)d_in[17];
    const float* o1b1   = (const float*)d_in[18];
    const float* o1W2   = (const float*)d_in[19];
    const float* o1b2   = (const float*)d_in[20];
    const float* o2W1   = (const float*)d_in[21];
    const float* o2b1   = (const float*)d_in[22];
    const float* o2W2   = (const float*)d_in[23];
    const float* o2b2   = (const float*)d_in[24];
    float* outp = (float*)d_out;

    // ---- workspace layout (floats, 64-element aligned), ints after
    float* W = (float*)d_ws;
    size_t o = 0;
    auto alloc = [&](size_t nelem) { size_t r = o; o += (nelem + 63) & ~(size_t)63; return r; };
    const size_t oH0  = alloc((size_t)NPTS * 64);
    const size_t oH1  = alloc(4096 * 64);
    const size_t oH2  = alloc(1024 * 64);
    const size_t oH3  = alloc(256 * 64);
    const size_t oP1  = alloc(4096 * 3);
    const size_t oN1  = alloc(4096 * 3);
    const size_t oP2  = alloc(1024 * 3);
    const size_t oN2  = alloc(1024 * 3);
    const size_t oP3  = alloc(256 * 3);
    const size_t oN3  = alloc(256 * 3);
    const size_t oDM  = alloc(NPTS);
    const size_t oCAT = alloc((size_t)NPTS * 128);
    const size_t oFA  = alloc((size_t)NPTS * 64);
    const size_t oFB  = alloc((size_t)NPTS * 64);
    int* ibase = (int*)(W + alloc(0));
    int* idxb  = ibase;              // 4096 ints
    int* nbb   = ibase + 4096;       // 4096*32 ints

    float* h0 = W + oH0; float* h1 = W + oH1; float* h2 = W + oH2; float* h3 = W + oH3;
    float* p1 = W + oP1; float* n1 = W + oN1;
    float* p2 = W + oP2; float* n2 = W + oN2;
    float* p3 = W + oP3; float* n3 = W + oN3;
    float* dmin = W + oDM; float* cat = W + oCAT;
    float* hfA = W + oFA; float* hfB = W + oFB;

    // ---- lin_in MLP
    lin_in_kernel<<<NPTS / 16, 128, 0, stream>>>(x, li_W1, li_b1, li_W2, li_b2, h0);

    // ---- SA levels
    const float* hsrc[3] = {h0, h1, h2};
    const float* psrc[3] = {pos, p1, p2};
    const float* nsrc[3] = {norm, n1, n2};
    float* hdst[3] = {h1, h2, h3};
    float* pdst[3] = {p1, p2, p3};
    float* ndst[3] = {n1, n2, n3};
    const int Nsrc[3] = {16384, 4096, 1024};
    const int Mq[3]   = {4096, 1024, 256};
    const float r2[3] = {0.01f, 0.04f, 0.16f};

    for (int i = 0; i < 3; ++i) {
        fps_kernel<<<1, 256, 0, stream>>>(psrc[i], Nsrc[i], Mq[i], idxb, dmin);
        gather_pn_kernel<<<(Mq[i] + 127) / 128, 128, 0, stream>>>(
            psrc[i], nsrc[i], idxb, Mq[i], pdst[i], ndst[i]);
        knn_radius_kernel<<<(Mq[i] + 63) / 64, 64, 0, stream>>>(
            pdst[i], Mq[i], psrc[i], Nsrc[i], r2[i], nbb);
        sa_kernel<<<Mq[i] / 2, 128, 0, stream>>>(
            hsrc[i], psrc[i], nsrc[i], pdst[i], ndst[i], Mq[i], nbb,
            sa1W1 + (size_t)i * D1 * D1, sa1b1 + (size_t)i * D1,
            sa1W2 + (size_t)i * D1 * D1, sa1b2 + (size_t)i * D1,
            sa2W + (size_t)i * D1 * NH, sa2b + (size_t)i * NH,
            hdst[i]);
    }

    // ---- FP levels (coarse -> fine)
    const float* hs_arr[3] = {h0, h1, h2};
    const float* ps_arr[3] = {pos, p1, p2};
    const int Nf_arr[3] = {16384, 4096, 1024};
    const float* hf = h3;
    const float* pc = p3;
    int Nc = 256;
    float* pp[2] = {hfA, hfB};
    int pb = 0;
    for (int i = 2; i >= 0; --i) {
        const int Nf = Nf_arr[i];
        fp_interp_kernel<<<(Nf + 127) / 128, 128, 0, stream>>>(
            ps_arr[i], Nf, pc, Nc, hf, hs_arr[i], cat);
        fp_mlp_kernel<<<Nf / 16, 128, 0, stream>>>(
            cat, fpW1 + (size_t)i * 128 * 128, fpb1 + (size_t)i * 128,
            fpW2 + (size_t)i * 128 * 64, fpb2 + (size_t)i * 64, pp[pb], Nf);
        hf = pp[pb]; pc = ps_arr[i]; Nc = Nf; pb ^= 1;
    }

    // ---- output heads
    head1_kernel<<<NPTS / 16, 128, 0, stream>>>(hf, o1W1, o1b1, o1W2, o1b2, outp);
    head2_kernel<<<1, 64, 0, stream>>>(h3, o2W1, o2b1, o2W2, o2b2, outp + (size_t)NPTS * 2);
}
